// ParallelESN_30872224923732
// MI455X (gfx1250) — matrix operand order repaired and checked by reading, not compiled, not run on hardware
//
#include <hip/hip_runtime.h>
#include <math.h>

// ---------------- problem constants (from reference setup) ----------------
#define Q      512
#define G      8
#define HID    4000
#define INSZ   76          // Q/G + 2*L = 64 + 12
#define LEAK   0.6f
#define STEPS  400
#define EPS    32
#define MT_PER_G 250       // HID / 16 row-tiles per group
#define KC_HH  125         // HID / 32 K-chunks for W_hh
#define KC_IN  3           // ceil(INSZ/32), padded to 96
#define NTILE  (G * MT_PER_G)  // 2000 row-tiles total

typedef __attribute__((ext_vector_type(16))) __bf16 v16bf;
typedef __attribute__((ext_vector_type(8)))  float  v8f;

__device__ __forceinline__ unsigned short f2bf(float f) {
  union { float f; unsigned u; } x; x.f = f;
  unsigned u = x.u;
  u += 0x7fffu + ((u >> 16) & 1u);      // round-to-nearest-even
  return (unsigned short)(u >> 16);
}
// unpack a u32 holding two packed bf16 into two floats
__device__ __forceinline__ void bf2x2(unsigned u, float& lo, float& hi) {
  union { unsigned u; float f; } a, b;
  a.u = u << 16; b.u = u & 0xffff0000u;
  lo = a.f; hi = b.f;
}

// ---------------------------------------------------------------------------
// Repack W_hh (f32 row-major [g][m][k]) into bf16 WMMA A-operand order:
// element i (0..15) of a lane's 32B payload maps to
//   K = i + (lane&16 ? 16 : 0),  M = lane & 15   (contiguous K order, matches B)
// Payloads stored contiguously as [tile][kc][lane][16].
// ---------------------------------------------------------------------------
__global__ __launch_bounds__(256) void pack_whh(const float* __restrict__ W,
                                                unsigned short* __restrict__ P) {
  long id = (long)blockIdx.x * 256 + threadIdx.x;        // one thread per lane-payload
  if (id >= (long)NTILE * KC_HH * 32) return;
  int  lane = (int)(id & 31);
  long t    = id >> 5;
  int  kc   = (int)(t % KC_HH);
  long tile = t / KC_HH;
  int  g    = (int)(tile / MT_PER_G);
  int  mt   = (int)(tile % MT_PER_G);
  int  M    = lane & 15;
  int  kb   = (lane & 16) ? 16 : 0;
  const float* src = W + ((size_t)g * HID + (size_t)mt * 16 + M) * HID + (size_t)kc * 32;
  unsigned short* dst = P + (size_t)id * 16;
  #pragma unroll
  for (int i = 0; i < 16; ++i) {
    int K = i + kb;
    dst[i] = f2bf(src[K]);
  }
}

// Same A-layout for W_in, K padded 76 -> 96 with zeros.
__global__ __launch_bounds__(256) void pack_win(const float* __restrict__ W,
                                                unsigned short* __restrict__ P) {
  int id = blockIdx.x * 256 + threadIdx.x;
  if (id >= NTILE * KC_IN * 32) return;
  int lane = id & 31;
  int t    = id >> 5;
  int kc   = t % KC_IN;
  int tile = t / KC_IN;
  int g    = tile / MT_PER_G;
  int mt   = tile % MT_PER_G;
  int M    = lane & 15;
  int kb   = (lane & 16) ? 16 : 0;
  const float* src = W + ((size_t)g * HID + (size_t)mt * 16 + M) * INSZ;
  unsigned short* dst = P + (size_t)id * 16;
  #pragma unroll
  for (int i = 0; i < 16; ++i) {
    int K = kc * 32 + i + kb;
    dst[i] = (K < INSZ) ? f2bf(src[K]) : (unsigned short)0;
  }
}

// W_out: plain f32 -> bf16 (row-major [g][o][8000]), halves readout traffic.
__global__ __launch_bounds__(256) void pack_wout(const float* __restrict__ W,
                                                 unsigned short* __restrict__ P) {
  int i = blockIdx.x * 256 + threadIdx.x;
  if (i < Q * 2 * HID) P[i] = f2bf(W[i]);
}

__global__ __launch_bounds__(256) void init_h(float* __restrict__ hf,
                                              unsigned short* __restrict__ hb) {
  int i = blockIdx.x * 256 + threadIdx.x;
  if (i < 2 * G * HID) { hf[i] = 0.0f; hb[i] = 0; }
}

// ---------------------------------------------------------------------------
// One ESN step: h_new = (1-LEAK)*h_old + LEAK*tanh(W_in*ug + W_hh*h_old)
// One wave per 16-row tile (2000 waves). B operand = h chunk replicated into
// all 16 columns (lanes 0-15 load K=k0..k0+15, lanes 16-31 K=k0+16..k0+31),
// so every column of D carries the matvec result. The periodic input gather
// is staged branch-free into LDS (zero-padded to 96 cols).
// ---------------------------------------------------------------------------
__global__ __launch_bounds__(256) void esn_update(
    const unsigned short* __restrict__ whh_p,
    const unsigned short* __restrict__ win_p,
    const float*          __restrict__ h_old,
    const unsigned short* __restrict__ hb_old,
    float*                __restrict__ h_new,
    unsigned short*       __restrict__ hb_new,
    const float*          __restrict__ u) {
  const int lane = threadIdx.x & 31;
  const int wave = threadIdx.x >> 5;
  const int tile = blockIdx.x * 8 + wave;          // 0..1999
  const int g    = tile / MT_PER_G;
  const int mt   = tile % MT_PER_G;
  const int koff = (lane & 16) ? 16 : 0;

  // ---- stage padded periodic-gather input table: lds_ug[g][0..95] ----
  __shared__ alignas(32) unsigned short lds_ug[G][96];
  for (int v = threadIdx.x; v < G * 96; v += 256) {
    int vg = v / 96, o = v % 96;
    lds_ug[vg][o] = (o < INSZ) ? f2bf(u[(vg * 64 - 6 + o) & (Q - 1)])
                               : (unsigned short)0;
  }
  __syncthreads();

  v8f c = {0.f, 0.f, 0.f, 0.f, 0.f, 0.f, 0.f, 0.f};

  // ---- input contribution: (16x96 padded W_in tile) x ug ----
  {
    const unsigned short* ap = win_p + (size_t)tile * KC_IN * 512 + (size_t)lane * 16;
    #pragma unroll
    for (int kc = 0; kc < KC_IN; ++kc) {
      v16bf a = *(const v16bf*)(ap + (size_t)kc * 512);
      v16bf b = *(const v16bf*)(&lds_ug[g][kc * 32 + koff]);   // ds_load_b128 x2
      c = __builtin_amdgcn_wmma_f32_16x16x32_bf16(false, a, false, b,
                                                  (short)0, c, false, false);
    }
  }

  // ---- recurrent contribution: stream the bf16 W_hh tile, chain WMMAs ----
  {
    const unsigned short* ap = whh_p + (size_t)tile * KC_HH * 512 + (size_t)lane * 16;
    const unsigned short* hb = hb_old + g * HID + koff;
    #pragma unroll 5
    for (int kc = 0; kc < KC_HH; ++kc) {
      __builtin_prefetch((const void*)(ap + 8 * 512), 0, 0);  // global_prefetch ahead
      v16bf a = *(const v16bf*)ap; ap += 512;
      v16bf b = *(const v16bf*)(hb + kc * 32);
      c = __builtin_amdgcn_wmma_f32_16x16x32_bf16(false, a, false, b,
                                                  (short)0, c, false, false);
    }
  }

  // All 16 columns identical: lane 0 holds M=0..7, lane 16 holds M=8..15.
  if ((lane & 15) == 0) {
    const int row0 = g * HID + mt * 16 + ((lane >> 4) << 3);
    #pragma unroll
    for (int r = 0; r < 8; ++r) {
      float pre = c[r];
      float hn = (1.0f - LEAK) * h_old[row0 + r] + LEAK * tanhf(pre);
      h_new[row0 + r]  = hn;
      hb_new[row0 + r] = f2bf(hn);                 // bf16 shadow for next step's B
    }
  }
}

// ---------------------------------------------------------------------------
// Readout: y[g,o] = sum_j Wout[g,o,j]*h[j] + Wout[g,o,HID+j]*h[j]^2
// One 256-thread block per output (512 blocks). 8 contiguous features per
// thread-iteration: b128 loads of packed bf16 weights + float4 loads of h.
// Writes the d_out row directly (which is also next step's input u).
// ---------------------------------------------------------------------------
__global__ __launch_bounds__(256) void esn_readout(
    const unsigned short* __restrict__ wout,
    const float*          __restrict__ h,
    float*                __restrict__ y) {
  const int go = blockIdx.x;                // g*64 + o
  const int g  = go >> 6;
  const unsigned short* wr = wout + (size_t)go * (2 * HID);
  const float* hg = h + g * HID;
  float s = 0.0f;
  for (int ch = threadIdx.x; ch < HID / 8; ch += 256) {
    int j = ch * 8;
    uint4  wa = *(const uint4*)(wr + j);          // 8 bf16, linear term
    uint4  wb = *(const uint4*)(wr + HID + j);    // 8 bf16, quadratic term
    float4 h0 = *(const float4*)(hg + j);
    float4 h1 = *(const float4*)(hg + j + 4);
    float a0, a1, a2, a3, a4, a5, a6, a7;
    float b0, b1, b2, b3, b4, b5, b6, b7;
    bf2x2(wa.x, a0, a1); bf2x2(wa.y, a2, a3);
    bf2x2(wa.z, a4, a5); bf2x2(wa.w, a6, a7);
    bf2x2(wb.x, b0, b1); bf2x2(wb.y, b2, b3);
    bf2x2(wb.z, b4, b5); bf2x2(wb.w, b6, b7);
    s += (a0 + b0 * h0.x) * h0.x + (a1 + b1 * h0.y) * h0.y;
    s += (a2 + b2 * h0.z) * h0.z + (a3 + b3 * h0.w) * h0.w;
    s += (a4 + b4 * h1.x) * h1.x + (a5 + b5 * h1.y) * h1.y;
    s += (a6 + b6 * h1.z) * h1.z + (a7 + b7 * h1.w) * h1.w;
  }
  #pragma unroll
  for (int off = 16; off > 0; off >>= 1) s += __shfl_down(s, off, 32);
  __shared__ float red[8];
  if ((threadIdx.x & 31) == 0) red[threadIdx.x >> 5] = s;
  __syncthreads();
  if (threadIdx.x == 0) {
    float t = 0.0f;
    #pragma unroll
    for (int w = 0; w < 8; ++w) t += red[w];
    y[go] = t;
  }
}

// ---------------------------------------------------------------------------
extern "C" void kernel_launch(void* const* d_in, const int* in_sizes, int n_in,
                              void* d_out, int out_size, void* d_ws, size_t ws_size,
                              hipStream_t stream) {
  (void)in_sizes; (void)n_in; (void)out_size; (void)ws_size;
  const float* u_hist = (const float*)d_in[0];   // [64, 512]
  const float* W_in   = (const float*)d_in[1];   // [8, 4000, 76]
  const float* W_hh   = (const float*)d_in[2];   // [8, 4000, 4000]
  const float* W_out  = (const float*)d_in[3];   // [8, 64, 8000]
  float* out = (float*)d_out;                    // [400, 512]

  // -------- workspace carve-up (~258 MiB; all 32B-aligned offsets) --------
  const size_t WHH_EL  = (size_t)NTILE * KC_HH * 512;  // 128,000,000 bf16
  const size_t WIN_EL  = (size_t)NTILE * KC_IN * 512;  //   3,072,000 bf16
  const size_t WOUT_EL = (size_t)Q * 2 * HID;          //   4,096,000 bf16
  unsigned short* whh_p  = (unsigned short*)d_ws;
  unsigned short* win_p  = whh_p + WHH_EL;
  unsigned short* wout_p = win_p + WIN_EL;
  float*          h_f32  = (float*)(wout_p + WOUT_EL); // 2 x [G*HID] ping-pong
  unsigned short* h_bf   = (unsigned short*)(h_f32 + 2 * G * HID);

  // -------- one-time (per launch) repack + state init --------
  {
    long n = (long)NTILE * KC_HH * 32;                  // 8,000,000
    pack_whh<<<(unsigned)((n + 255) / 256), 256, 0, stream>>>(W_hh, whh_p);
  }
  pack_win <<<(NTILE * KC_IN * 32 + 255) / 256, 256, 0, stream>>>(W_in, win_p);
  pack_wout<<<(Q * 2 * HID + 255) / 256,       256, 0, stream>>>(W_out, wout_p);
  init_h   <<<(2 * G * HID + 255) / 256,       256, 0, stream>>>(h_f32, h_bf);

  const int HN = G * HID;  // 32000

  // -------- warmup: 32 teacher-forced steps over u_hist[32..63] --------
  for (int t = 0; t < EPS; ++t) {
    int p = t & 1;
    esn_update<<<MT_PER_G, 256, 0, stream>>>(
        whh_p, win_p,
        h_f32 + p * HN,        h_bf + p * HN,
        h_f32 + (p ^ 1) * HN,  h_bf + (p ^ 1) * HN,
        u_hist + (size_t)(64 - EPS + t) * Q);
  }

  // -------- closed loop: 400 predict steps; d_out rows are the feedback --------
  for (int s = 0; s < STEPS; ++s) {
    int t = EPS + s;
    int p = t & 1;
    const float* u = (s == 0) ? (u_hist + (size_t)63 * Q)
                              : (out + (size_t)(s - 1) * Q);
    esn_update<<<MT_PER_G, 256, 0, stream>>>(
        whh_p, win_p,
        h_f32 + p * HN,        h_bf + p * HN,
        h_f32 + (p ^ 1) * HN,  h_bf + (p ^ 1) * HN, u);
    esn_readout<<<Q, 256, 0, stream>>>(wout_p, h_f32 + (p ^ 1) * HN,
                                       out + (size_t)s * Q);
  }
}
